// PulseLoss_29454885716315
// MI455X (gfx1250) — matrix operand order repaired and checked by reading, not compiled, not run on hardware
//
#include <hip/hip_runtime.h>

typedef __attribute__((ext_vector_type(16))) _Float16 v16h;
typedef __attribute__((ext_vector_type(8)))  float    v8f;

constexpr int   BATCH  = 32;
constexpr int   SEQ    = 8192;
constexpr int   TILE_N = 16;             // seq positions per WMMA tile (N)
constexpr int   KOFF   = 24;             // window left extent: t in [s0-24, s0+39]
constexpr int   WAVES_PER_BLOCK = 8;     // 256 threads, wave32
constexpr int   NTILE_B = BATCH / 16;    // 2
constexpr int   NJOBS   = (SEQ / TILE_N) * NTILE_B;  // 1024 wave-jobs
constexpr int   NBLOCKS = NJOBS / WAVES_PER_BLOCK;   // 128
constexpr float LN2     = 0.69314718055994530942f;
constexpr float LOG2E   = 1.44269504088896340736f;

// Fused kernel: banded gaussian-mask matmul via v_wmma_f32_16x16x32_f16,
// clip, BCE in log2 domain (hardware v_log_f32), block partial sums.
__global__ __launch_bounds__(256)
void pulse_loss_main(const float* __restrict__ pred,
                     const int*   __restrict__ target,
                     float*       __restrict__ partial)
{
    __shared__ float smem[WAVES_PER_BLOCK];

    const int lane = threadIdx.x & 31;
    const int wave = threadIdx.x >> 5;
    const int job  = blockIdx.x * WAVES_PER_BLOCK + wave;   // < 1024
    const int bh   = job & (NTILE_B - 1);                   // batch half (0/1)
    const int s0   = (job >> 1) * TILE_N;                   // tile start position

    const int n    = lane & 15;   // N column (seq offset in tile); also M row for A
    const int half = lane >> 4;

    // ---- Constant Gaussian band operand B (K x N), built once per wave.
    // B[k][n] = exp(-d*d/8) = exp2(-d*d/8 * log2(e)), d = k - n - KOFF.
    // B f16 layout: element i -> K = i + 8*((i>>3) + half) within each 32-deep
    // matrix (matches A: K 0-7/16-23 in lanes 0-15, 8-15/24-31 in lanes 16-31).
    v16h b0, b1;
#pragma unroll
    for (int i = 0; i < 16; ++i) {
        const int   kl = i + 8 * ((i >> 3) + half);
        const float d0 = (float)(kl - n - KOFF);
        const float d1 = d0 + 32.0f;
        b0[i] = (_Float16)__builtin_amdgcn_exp2f(d0 * d0 * (-0.125f * LOG2E));
        b1[i] = (_Float16)__builtin_amdgcn_exp2f(d1 * d1 * (-0.125f * LOG2E));
    }

    // ---- A operand: 0/1 peak mask window (exact in f16).
    // A f16 16x32 layout: lane m = lane&15 (both halves);
    // element i -> K_local = i + 8*((i>>3) + half), i.e. four 8-consecutive-t
    // groups starting at g0 = s0-KOFF+8*half, g0+16, g0+32, g0+48.
    const long trow = (long)(bh * 16 + n) * SEQ;
    const int  g0   = s0 - KOFF + 8 * half;
    int mw[4][8];

    // Wave-uniform interior test (scalar branch; EXEC stays all-1s for WMMA).
    if (s0 >= KOFF && s0 <= SEQ - 40) {
        // Group starts are 8-int (32B) aligned: two aligned int4 loads each.
#pragma unroll
        for (int gidx = 0; gidx < 4; ++gidx) {
            const int4* p = (const int4*)(target + trow + g0 + 16 * gidx);
            const int4  x = p[0];
            const int4  y = p[1];
            mw[gidx][0] = x.x; mw[gidx][1] = x.y; mw[gidx][2] = x.z; mw[gidx][3] = x.w;
            mw[gidx][4] = y.x; mw[gidx][5] = y.y; mw[gidx][6] = y.z; mw[gidx][7] = y.w;
        }
    } else {
        // Boundary tiles: clamped scalar loads, OOB forced to zero.
#pragma unroll
        for (int gidx = 0; gidx < 4; ++gidx) {
#pragma unroll
            for (int j = 0; j < 8; ++j) {
                const int t = g0 + 16 * gidx + j;
                const int c = min(max(t, 0), SEQ - 1);
                const int v = target[trow + c];
                mw[gidx][j] = ((t >= 0) & (t < SEQ)) ? v : 0;
            }
        }
    }

    v16h a0, a1;
#pragma unroll
    for (int i = 0; i < 8; ++i) {
        a0[i]     = (_Float16)((mw[0][i] != 0) ? 1.0f : 0.0f);  // K_local = i + 8*half
        a0[i + 8] = (_Float16)((mw[1][i] != 0) ? 1.0f : 0.0f);  // K_local = i+16 + 8*half
        a1[i]     = (_Float16)((mw[2][i] != 0) ? 1.0f : 0.0f);
        a1[i + 8] = (_Float16)((mw[3][i] != 0) ? 1.0f : 0.0f);
    }

    // ---- D = A0*B0 + A1*B1 : gaussian_target for 16 batches x 16 positions.
    v8f acc = {0.f, 0.f, 0.f, 0.f, 0.f, 0.f, 0.f, 0.f};
    acc = __builtin_amdgcn_wmma_f32_16x16x32_f16(false, a0, false, b0,
                                                 (short)0, acc, false, false);
    acc = __builtin_amdgcn_wmma_f32_16x16x32_f16(false, a1, false, b1,
                                                 (short)0, acc, false, false);

    // D layout: acc[r] -> batch 16*bh + r + 8*half, seq s0 + n.
    // BCE accumulated in log2 domain; scaled by ln2 once at the end.
    const float* prow = pred + (long)(bh * 16 + 8 * half) * SEQ + (s0 + n);
    float local = 0.0f;
#pragma unroll
    for (int r = 0; r < 8; ++r) {
        float g = fminf(fmaxf(acc[r], 0.0f), 1.0f);      // jnp.clip(., 0, 1)
        const float p  = prow[(long)r * SEQ];
        const float lp = __log2f(p);                     // v_log_f32
        const float lq = __log2f(1.0f - p);              // p in [.01,.99]: safe
        local = fmaf(g, lp - lq, local + lq);            // g*lp + (1-g)*lq
    }

    // Wave32 reduction, then cross-wave via LDS.
#pragma unroll
    for (int off = 16; off >= 1; off >>= 1)
        local += __shfl_xor(local, off, 32);
    if (lane == 0) smem[wave] = local;
    __syncthreads();
    if (threadIdx.x == 0) {
        float acc_b = 0.0f;
#pragma unroll
        for (int w = 0; w < WAVES_PER_BLOCK; ++w) acc_b += smem[w];
        partial[blockIdx.x] = acc_b;
    }
}

// Deterministic final fold: ordered sum of 128 partials -> mean BCE (ln2 scale).
__global__ void pulse_loss_finish(const float* __restrict__ partial,
                                  float* __restrict__ out)
{
    if (threadIdx.x == 0 && blockIdx.x == 0) {
        float total = 0.0f;
        for (int i = 0; i < NBLOCKS; ++i) total += partial[i];
        out[0] = -total * (LN2 / (float)(BATCH * SEQ));
    }
}

extern "C" void kernel_launch(void* const* d_in, const int* in_sizes, int n_in,
                              void* d_out, int out_size, void* d_ws, size_t ws_size,
                              hipStream_t stream) {
    (void)in_sizes; (void)n_in; (void)out_size; (void)ws_size;
    const float* pred   = (const float*)d_in[0];
    const int*   target = (const int*)d_in[1];
    float*       out    = (float*)d_out;
    float*       parts  = (float*)d_ws;   // 128 floats of scratch

    pulse_loss_main<<<NBLOCKS, WAVES_PER_BLOCK * 32, 0, stream>>>(pred, target, parts);
    pulse_loss_finish<<<1, 32, 0, stream>>>(parts, out);
}
